// PluginLstmRnntPost_83056077570358
// MI455X (gfx1250) — compile-verified, matrix operand order checked by reading, not run on hardware
//
#include <hip/hip_runtime.h>
#include <hip/hip_bf16.h>

// ---------------- problem constants (match reference) ----------------
#define TT    256
#define BB    64
#define INDIM 2048
#define HH    1024
#define GG    (4 * HH)      // 4096 gate width
#define NL    3

typedef __attribute__((ext_vector_type(16))) __bf16 v16bf;
typedef __attribute__((ext_vector_type(8)))  __bf16 v8bf;
typedef __attribute__((ext_vector_type(8)))  float  v8f;
typedef int v4i_ __attribute__((vector_size(16)));
typedef __attribute__((address_space(1))) v4i_* gas_v4i_ptr;   // global
typedef __attribute__((address_space(3))) v4i_* las_v4i_ptr;   // LDS

#if __has_builtin(__builtin_amdgcn_global_load_async_to_lds_b128)
#define HAVE_ASYNC_LDS 1
#else
#define HAVE_ASYNC_LDS 0
#endif

// ---------------------------------------------------------------------
// Fragment loader following the CDNA5 16-bit A-matrix 16x32 layout:
// lanes 0-15 hold K = [0..7] and [16..23]; lanes 16-31 hold K = [8..15]
// and [24..31].  Per lane: two contiguous 8-element (16B) chunks.
// Works for global or LDS-resident sources (compiler picks b128/ds_b128).
// ---------------------------------------------------------------------
__device__ __forceinline__ v16bf load_frag(const __bf16* __restrict__ base,
                                           int row, int ld, int k0, int lane) {
    const int hi = lane >> 4;                       // 0 or 1
    const __bf16* p = base + (size_t)row * ld + k0 + (hi << 3);
    union { v16bf v; v8bf h[2]; } u;
    u.h[0] = *(const v8bf*)(p);
    u.h[1] = *(const v8bf*)(p + 16);
    return u.v;
}

__device__ __forceinline__ float sigmoidf_(float x) {
    return 1.0f / (1.0f + __expf(-x));
}

// ---------------------------------------------------------------------
// fp32 -> bf16 conversion
// ---------------------------------------------------------------------
__global__ void __launch_bounds__(256)
cvt_f32_bf16(const float* __restrict__ src, __bf16* __restrict__ dst, long n) {
    long i = (long)blockIdx.x * blockDim.x + threadIdx.x;
    if (i < n) dst[i] = (__bf16)src[i];
}

// ---------------------------------------------------------------------
// xg = X(bf16)[M,K] * W(bf16)[N,K]^T + (b_ih + b_hh)   -> XG fp32 [M,N]
// One wave per 16x64 output strip: one A fragment feeds 4 WMMAs.
// ---------------------------------------------------------------------
__global__ void __launch_bounds__(128)
lstm_gemm_xg(const __bf16* __restrict__ X, const __bf16* __restrict__ W,
             const float* __restrict__ b_ih, const float* __restrict__ b_hh,
             float* __restrict__ XG, int M, int N, int K) {
    const int lane    = threadIdx.x & 31;
    const int waveId  = blockIdx.x * (blockDim.x >> 5) + (threadIdx.x >> 5);
    const int stripsN = N >> 6;                 // strips of 64 columns
    const int tm = waveId / stripsN;
    const int tn = waveId - tm * stripsN;
    const int m0 = tm << 4, n0 = tn << 6;
    const int ar = m0 + (lane & 15);
    const int l16 = lane & 15;

    v8f acc[4] = {};
    #pragma unroll 2
    for (int k0 = 0; k0 < K; k0 += 32) {
        __builtin_prefetch(X + (size_t)ar * K + k0 + 64, 0, 1);     // next K-block
        v16bf a = load_frag(X, ar, K, k0, lane);
        #pragma unroll
        for (int j = 0; j < 4; ++j) {
            v16bf b = load_frag(W, n0 + (j << 4) + l16, K, k0, lane);
            acc[j] = __builtin_amdgcn_wmma_f32_16x16x32_bf16(
                         false, a, false, b, (short)0, acc[j], false, false);
        }
    }

    const int rbase = m0 + ((lane >> 4) << 3);
    #pragma unroll
    for (int j = 0; j < 4; ++j) {
        const int col = n0 + (j << 4) + l16;
        const float bias = b_ih[col] + b_hh[col];
        float* out = XG + (size_t)rbase * N + col;
        #pragma unroll
        for (int r = 0; r < 8; ++r) out[(size_t)r * N] = acc[j][r] + bias;
    }
}

// ---------------------------------------------------------------------
// One recurrent timestep.  Block = 128 threads = 4 waves, one per gate.
// The 16x1024 bf16 h-tile (shared by all four gate GEMMs) is staged into
// LDS once per block via the CDNA5 async global->LDS path (ASYNCcnt),
// then each wave reads A fragments with ds_load while streaming its
// gate's Whh rows from global/L2.
// ---------------------------------------------------------------------
__global__ void __launch_bounds__(128)
lstm_step(const float*  __restrict__ XGt,   // [B, 4H] slice at time t
          const __bf16* __restrict__ Hin,   // [B, H]  h_{t-1} (bf16)
          const __bf16* __restrict__ Whh,   // [4H, H] (bf16)
          float*        __restrict__ C,     // [B, H]  cell state (in/out)
          __bf16*       __restrict__ Hout,  // [B, H]  h_t (bf16)
          float*        __restrict__ Y,     // [B, H]  h_t fp32 (last layer) or null
          float*        __restrict__ Hfin) {// [B, H]  final-h slot or null
    __shared__ __bf16 aTile[16 * HH];       // 32 KB h-tile
    __shared__ float  gt[4][16][17];        // gate exchange (+1 pad)

    const int lane = threadIdx.x & 31;
    const int gate = threadIdx.x >> 5;      // 0..3  (i, f, g, o)
    const int tn   = blockIdx.x & (HH / 16 - 1);
    const int tm   = blockIdx.x >> 6;       // B/16 = 4 tiles
    const int m0   = tm << 4;
    const int n0   = tn << 4;
    const int br   = gate * HH + n0 + (lane & 15);

    // ---- stage the 16 x HH bf16 A-tile into LDS (32768 bytes) ----
    {
        const char* gsrc = (const char*)(Hin + (size_t)m0 * HH);
        char*       ldst = (char*)aTile;
        int ofs = (int)threadIdx.x * 16;
        #pragma unroll
        for (int it = 0; it < 16; ++it, ofs += 128 * 16) {
#if HAVE_ASYNC_LDS
            __builtin_amdgcn_global_load_async_to_lds_b128(
                (gas_v4i_ptr)(gsrc + ofs), (las_v4i_ptr)(ldst + ofs), 0, 0);
#else
            *(v8bf*)(ldst + ofs) = *(const v8bf*)(gsrc + ofs);
#endif
        }
#if HAVE_ASYNC_LDS
#if __has_builtin(__builtin_amdgcn_s_wait_asynccnt)
        __builtin_amdgcn_s_wait_asynccnt(0);
#else
        asm volatile("s_wait_asynccnt 0x0" ::: "memory");
#endif
#endif
    }
    __syncthreads();

    v8f acc = {};
    #pragma unroll 8
    for (int k0 = 0; k0 < HH; k0 += 32) {
        v16bf a = load_frag(aTile, lane & 15, HH, k0, lane);   // ds_load_b128 x2
        v16bf b = load_frag(Whh, br, HH, k0, lane);            // global_load_b128 x2
        acc = __builtin_amdgcn_wmma_f32_16x16x32_bf16(
                  false, a, false, b, (short)0, acc, false, false);
    }

    const int l16  = lane & 15;
    const int lhi  = lane >> 4;
    const int col4 = gate * HH + n0 + l16;
    #pragma unroll
    for (int r = 0; r < 8; ++r) {
        const int lm = r + (lhi << 3);
        gt[gate][lm][l16] = acc[r] + XGt[(size_t)(m0 + lm) * GG + col4];
    }
    __syncthreads();

    #pragma unroll
    for (int idx = threadIdx.x; idx < 256; idx += 128) {
        const int lm = idx >> 4, ln = idx & 15;
        const size_t off = (size_t)(m0 + lm) * HH + (n0 + ln);
        const float i = sigmoidf_(gt[0][lm][ln]);
        const float f = sigmoidf_(gt[1][lm][ln]);
        const float g = tanhf(gt[2][lm][ln]);
        const float o = sigmoidf_(gt[3][lm][ln]);
        const float c = f * C[off] + i * g;
        C[off] = c;
        const float h = o * tanhf(c);
        Hout[off] = (__bf16)h;
        if (Y)    Y[off] = h;
        if (Hfin) Hfin[off] = h;
    }
}

// ---------------------------------------------------------------------
// Host-side orchestration
// ---------------------------------------------------------------------
extern "C" void kernel_launch(void* const* d_in, const int* in_sizes, int n_in,
                              void* d_out, int out_size, void* d_ws, size_t ws_size,
                              hipStream_t stream) {
    (void)in_sizes; (void)n_in; (void)out_size; (void)ws_size;

    const float* x  = (const float*)d_in[0];
    const float* h0 = (const float*)d_in[1];
    const float* c0 = (const float*)d_in[2];
    const float* wih[NL] = {(const float*)d_in[3], (const float*)d_in[7],  (const float*)d_in[11]};
    const float* whh[NL] = {(const float*)d_in[4], (const float*)d_in[8],  (const float*)d_in[12]};
    const float* bih[NL] = {(const float*)d_in[5], (const float*)d_in[9],  (const float*)d_in[13]};
    const float* bhh[NL] = {(const float*)d_in[6], (const float*)d_in[10], (const float*)d_in[14]};

    float* out   = (float*)d_out;
    float* y_out = out;                                   // [T,B,H]
    float* h_out = out + (size_t)TT * BB * HH;            // [3,B,H]
    float* c_out = h_out + (size_t)NL * BB * HH;          // [3,B,H]

    // ---- workspace carve-up (256B aligned) ----
    char* ws = (char*)d_ws;
    auto carve = [&](size_t bytes) -> char* {
        char* p = ws; ws += (bytes + 255) & ~(size_t)255; return p;
    };
    __bf16* xbf = (__bf16*)carve((size_t)TT * BB * INDIM * 2);
    __bf16* wihbf[NL];
    __bf16* whhbf[NL];
    for (int l = 0; l < NL; ++l) {
        const size_t kin = (l == 0) ? INDIM : HH;
        wihbf[l] = (__bf16*)carve((size_t)GG * kin * 2);
        whhbf[l] = (__bf16*)carve((size_t)GG * HH * 2);
    }
    __bf16* hseq[2];
    hseq[0] = (__bf16*)carve((size_t)(TT + 1) * BB * HH * 2);
    hseq[1] = (__bf16*)carve((size_t)(TT + 1) * BB * HH * 2);
    float* xg = (float*)carve((size_t)TT * BB * GG * 4);

    // ---- conversions (every call; deterministic) ----
    {
        long n = (long)TT * BB * INDIM;
        cvt_f32_bf16<<<(unsigned)((n + 255) / 256), 256, 0, stream>>>(x, xbf, n);
    }
    for (int l = 0; l < NL; ++l) {
        const long kin = (l == 0) ? INDIM : HH;
        long n1 = (long)GG * kin;
        cvt_f32_bf16<<<(unsigned)((n1 + 255) / 256), 256, 0, stream>>>(wih[l], wihbf[l], n1);
        long n2 = (long)GG * HH;
        cvt_f32_bf16<<<(unsigned)((n2 + 255) / 256), 256, 0, stream>>>(whh[l], whhbf[l], n2);
    }
    // cell state lives in d_out's c slot; re-init from c0 each call
    (void)hipMemcpyAsync(c_out, c0, (size_t)NL * BB * HH * sizeof(float),
                         hipMemcpyDeviceToDevice, stream);

    // ---- layer pipeline ----
    const __bf16* layerX = xbf;
    int K = INDIM;
    for (int l = 0; l < NL; ++l) {
        __bf16* hs = hseq[l & 1];
        // slice 0 of the hidden sequence = bf16(h0[l])
        {
            long n = (long)BB * HH;
            cvt_f32_bf16<<<(unsigned)((n + 255) / 256), 256, 0, stream>>>(
                h0 + (size_t)l * BB * HH, hs, n);
        }
        // big parallel GEMM: xg[t,b,:] = X W_ih^T + (b_ih + b_hh)
        {
            const int M = TT * BB;                        // 16384
            const int waves = (M / 16) * (GG / 64);       // 16x64 strips
            lstm_gemm_xg<<<waves / 4, 128, 0, stream>>>(
                layerX, wihbf[l], bih[l], bhh[l], xg, M, GG, K);
        }
        // sequential scan: 256 dependent step kernels
        for (int t = 0; t < TT; ++t) {
            lstm_step<<<(BB / 16) * (HH / 16), 128, 0, stream>>>(
                xg + (size_t)t * BB * GG,
                hs + (size_t)t * BB * HH,
                whhbf[l],
                c_out + (size_t)l * BB * HH,
                hs + (size_t)(t + 1) * BB * HH,
                (l == NL - 1) ? (y_out + (size_t)t * BB * HH) : nullptr,
                (t == TT - 1) ? (h_out + (size_t)l * BB * HH) : nullptr);
        }
        layerX = hs + (size_t)BB * HH;   // slices 1..T feed the next layer
        K = HH;
    }
}
